// HGTLayer_81587198755412
// MI455X (gfx1250) — matrix operand order repaired and checked by reading, not compiled, not run on hardware
//
#include <hip/hip_runtime.h>
#include <math.h>

// ---------------------------------------------------------------------------
// HGT layer forward for MI455X (gfx1250, wave32, WMMA).
//   N=200000 nodes/type, E=1000000 edges/type, IN=OUT=128, H=4, DK=32.
// All GEMM-shaped math runs on v_wmma_f32_16x16x32_bf16 (f32 accumulate).
// Pipeline (all on `stream`, stream order provides the inter-stage barriers):
//   0) zero agg/den/segmax
//   1) proj:      K/Q/M = x @ W{k,q,m} + b  -> bf16       (WMMA)
//   2) edge_att:  att = ((K[src]@w_att) . Q[dst]) * mu/sqrt(DK); atomicMax segmax (WMMA)
//   3) edge_agg:  w = exp(att - segmax); den += w; agg += w * (M[src]@w_msg)   (WMMA)
//   4) gelu:      G = bf16(gelu(agg / (den + 1e-16)))
//   5) out:       LN(alpha*(G@Wa + ba) + (1-alpha)*x)     (WMMA + shfl LN)
// Workspace requirement: ~660 MB.
// ---------------------------------------------------------------------------

typedef __attribute__((ext_vector_type(16))) __bf16 v16bf;
typedef __attribute__((ext_vector_type(8)))  __bf16 v8bf;
typedef __attribute__((ext_vector_type(8)))  float  v8f;

#define NN 200000
#define EE 1000000
#define HH 4
#define DKK 32
#define FF 128

static __device__ __forceinline__ v16bf cat8(v8bf lo, v8bf hi) {
  return __builtin_shufflevector(lo, hi, 0,1,2,3,4,5,6,7,8,9,10,11,12,13,14,15);
}

// A fragment (16x32 bf16): lane holds row M = lane&15; k = base+{0..7,16..23},
// base = (lane>>4)*8.  p must already point at row_ptr + base.
static __device__ __forceinline__ v16bf a_frag_bf16(const __bf16* p) {
  v8bf lo = *reinterpret_cast<const v8bf*>(p);        // k = base+0..7   (16B)
  v8bf hi = *reinterpret_cast<const v8bf*>(p + 16);   // k = base+16..23 (16B)
  return cat8(lo, hi);
}

// Same A fragment but converting from an f32 row on the fly.
static __device__ __forceinline__ v16bf a_frag_f32(const float* p) {
  v16bf r;
#pragma unroll
  for (int i = 0; i < 8; ++i) r[i] = (__bf16)p[i];
#pragma unroll
  for (int i = 0; i < 8; ++i) r[8 + i] = (__bf16)p[16 + i];
  return r;
}

// B fragment (32x16 bf16): lane holds column f = lane&15; k = base+{0..7,16..23},
// base = (lane>>4)*8.  p points at &wT[f*K + k0 + base] where wT is W transposed [f][k] in LDS.
static __device__ __forceinline__ v16bf b_frag(const __bf16* p) {
  v8bf lo = *reinterpret_cast<const v8bf*>(p);
  v8bf hi = *reinterpret_cast<const v8bf*>(p + 16);
  return cat8(lo, hi);
}

static __device__ __forceinline__ v8f wmma_bf16(v16bf a, v16bf b, v8f c) {
  return __builtin_amdgcn_wmma_f32_16x16x32_bf16(false, a, false, b, (short)0, c,
                                                 false, false);
}

// ---------------------------------------------------------------------------
__global__ void hgt_zero(uint4* __restrict__ p, long long n16) {
  long long i = (long long)blockIdx.x * blockDim.x + threadIdx.x;
  const long long st = (long long)gridDim.x * blockDim.x;
  uint4 z; z.x = z.y = z.z = z.w = 0u;
  for (; i < n16; i += st) p[i] = z;
}

// ---------------------------------------------------------------------------
// K/Q/M projections: per node type, 64 rows per block, 8 waves = 8 column tiles.
__global__ __launch_bounds__(256) void hgt_proj(
    const float* __restrict__ x,
    const float* __restrict__ Wk, const float* __restrict__ bk,
    const float* __restrict__ Wq, const float* __restrict__ bq,
    const float* __restrict__ Wm, const float* __restrict__ bm,
    __bf16* __restrict__ Kb, __bf16* __restrict__ Qb, __bf16* __restrict__ Mb) {
  __shared__ __bf16 wT[FF * FF];                      // 32 KB, transposed [f][k]
  const int t = blockIdx.y;
  const int rowBase = blockIdx.x * 64;
  const int wv = threadIdx.x >> 5;
  const int lane = threadIdx.x & 31;
  const int hl = lane >> 4, l15 = lane & 15;
  const int kA = hl * 8;
  const int fcol = wv * 16 + l15;

  const float* Ws[3] = {Wk + (size_t)t * FF * FF, Wq + (size_t)t * FF * FF,
                        Wm + (size_t)t * FF * FF};
  const float* Bs[3] = {bk + t * FF, bq + t * FF, bm + t * FF};
  __bf16* Os[3] = {Kb + (size_t)t * NN * FF, Qb + (size_t)t * NN * FF,
                   Mb + (size_t)t * NN * FF};
  const float* xrow0 = x + ((size_t)t * NN + rowBase) * FF;

  for (int mat = 0; mat < 3; ++mat) {
    __syncthreads();                                  // WAR guard on wT
    for (int idx = threadIdx.x; idx < FF * FF; idx += 256) {
      const int k = idx >> 7, f = idx & 127;
      wT[f * FF + k] = (__bf16)Ws[mat][idx];
    }
    __syncthreads();
    v16bf Bf[4];
#pragma unroll
    for (int ks = 0; ks < 4; ++ks)
      Bf[ks] = b_frag(&wT[fcol * FF + ks * 32 + hl * 8]);
    const float bias = Bs[mat][fcol];
#pragma unroll
    for (int rt = 0; rt < 4; ++rt) {
      const float* ar = xrow0 + (size_t)(rt * 16 + l15) * FF;
      v8f acc = {0.f, 0.f, 0.f, 0.f, 0.f, 0.f, 0.f, 0.f};
#pragma unroll
      for (int ks = 0; ks < 4; ++ks)
        acc = wmma_bf16(a_frag_f32(ar + ks * 32 + kA), Bf[ks], acc);
      __bf16* op = Os[mat] + (size_t)(rowBase + rt * 16) * FF + fcol;
#pragma unroll
      for (int j = 0; j < 8; ++j)
        op[(size_t)(j + hl * 8) * FF] = (__bf16)(acc[j] + bias);
    }
  }
}

// ---------------------------------------------------------------------------
// Edge attention: one (edge-type, head) per blockIdx.{z,y}; 16 edges per wave-group.
__global__ __launch_bounds__(256) void hgt_edge_att(
    const __bf16* __restrict__ Kb, const __bf16* __restrict__ Qb,
    const int* __restrict__ eidx, const float* __restrict__ w_att,
    const float* __restrict__ mu, float* __restrict__ att,
    unsigned int* __restrict__ segmax) {
  const int h = blockIdx.y;
  const int t = blockIdx.z;
  const int s = t, d = 1 - t;                          // EDGE_META = ((0,1),(1,0))
  __shared__ __bf16 wT[DKK * DKK];                     // transposed [f][k], 2 KB
  for (int idx = threadIdx.x; idx < DKK * DKK; idx += 256) {
    const int k = idx >> 5, f = idx & 31;
    wT[f * DKK + k] = (__bf16)w_att[(((size_t)t * HH + h) * DKK + k) * DKK + f];
  }
  __syncthreads();
  const int wv = threadIdx.x >> 5, lane = threadIdx.x & 31;
  const int hl = lane >> 4, l15 = lane & 15;
  const v16bf B0 = b_frag(&wT[l15 * DKK + hl * 8]);          // f = 0..15
  const v16bf B1 = b_frag(&wT[(16 + l15) * DKK + hl * 8]);   // f = 16..31
  const float ms = mu[t * HH + h] * 0.17677669529663687f;     // mu / sqrt(DK)
  const int* srcA = eidx + ((size_t)t * 2 + 0) * EE;
  const int* dstA = eidx + ((size_t)t * 2 + 1) * EE;
  const __bf16* Kp = Kb + (size_t)s * NN * FF;
  const __bf16* Qp = Qb + (size_t)d * NN * FF;
  float* attT = att + (size_t)t * EE * HH;
  unsigned int* smx = segmax + (size_t)d * NN * HH;
  const long long gw = (long long)blockIdx.x * 8 + wv;
  const long long ngw = (long long)gridDim.x * 8;
  for (long long e0 = gw * 16; e0 < EE; e0 += ngw * 16) {
    const int src = srcA[e0 + l15];
    const int dst = dstA[e0 + l15];
    const v16bf A = a_frag_bf16(Kp + (size_t)src * FF + h * DKK + hl * 8);
    const v8f z = {0.f, 0.f, 0.f, 0.f, 0.f, 0.f, 0.f, 0.f};
    const v8f c0 = wmma_bf16(A, B0, z);                // k_tr, f tile 0
    const v8f c1 = wmma_bf16(A, B1, z);                // k_tr, f tile 1
#pragma unroll
    for (int j = 0; j < 8; ++j) {
      const int m = j + hl * 8;                        // edge index within group
      const int dm = __shfl(dst, m, 32);
      const __bf16* qrow = Qp + (size_t)dm * FF + h * DKK;
      float p = c0[j] * (float)qrow[l15] + c1[j] * (float)qrow[16 + l15];
      p += __shfl_xor(p, 1, 32);                       // reduce over 16 f-lanes
      p += __shfl_xor(p, 2, 32);
      p += __shfl_xor(p, 4, 32);
      p += __shfl_xor(p, 8, 32);
      if (l15 == 0) {
        const float a = p * ms;
        attT[(size_t)(e0 + m) * HH + h] = a;
        const unsigned int bits = __float_as_uint(a);  // order-preserving key
        const unsigned int key =
            (bits & 0x80000000u) ? ~bits : (bits | 0x80000000u);
        atomicMax(&smx[(size_t)dm * HH + h], key);
      }
    }
  }
}

// ---------------------------------------------------------------------------
// Edge aggregation: w = exp(att - segmax[dst]); den += w; agg += w * (M[src] @ w_msg).
__global__ __launch_bounds__(256) void hgt_edge_agg(
    const __bf16* __restrict__ Mb, const int* __restrict__ eidx,
    const float* __restrict__ w_msg, const float* __restrict__ att,
    const unsigned int* __restrict__ segmax, float* __restrict__ den,
    float* __restrict__ agg) {
  const int h = blockIdx.y;
  const int t = blockIdx.z;
  const int s = t, d = 1 - t;
  __shared__ __bf16 wT[DKK * DKK];
  for (int idx = threadIdx.x; idx < DKK * DKK; idx += 256) {
    const int k = idx >> 5, f = idx & 31;
    wT[f * DKK + k] = (__bf16)w_msg[(((size_t)t * HH + h) * DKK + k) * DKK + f];
  }
  __syncthreads();
  const int wv = threadIdx.x >> 5, lane = threadIdx.x & 31;
  const int hl = lane >> 4, l15 = lane & 15;
  const v16bf B0 = b_frag(&wT[l15 * DKK + hl * 8]);
  const v16bf B1 = b_frag(&wT[(16 + l15) * DKK + hl * 8]);
  const int* srcA = eidx + ((size_t)t * 2 + 0) * EE;
  const int* dstA = eidx + ((size_t)t * 2 + 1) * EE;
  const __bf16* Mp = Mb + (size_t)s * NN * FF;
  const float* attT = att + (size_t)t * EE * HH;
  const unsigned int* smx = segmax + (size_t)d * NN * HH;
  float* denp = den + (size_t)d * NN * HH;
  float* aggp = agg + (size_t)d * NN * FF;
  const long long gw = (long long)blockIdx.x * 8 + wv;
  const long long ngw = (long long)gridDim.x * 8;
  for (long long e0 = gw * 16; e0 < EE; e0 += ngw * 16) {
    const int src = srcA[e0 + l15];
    const int dst = dstA[e0 + l15];
    const float av = attT[(size_t)(e0 + l15) * HH + h];
    const unsigned int key = smx[(size_t)dst * HH + h];
    const unsigned int bits = (key & 0x80000000u) ? (key & 0x7FFFFFFFu) : ~key;
    const float w = expf(av - __uint_as_float(bits));
    if (hl == 0) atomicAdd(&denp[(size_t)dst * HH + h], w);
    const v16bf A = a_frag_bf16(Mp + (size_t)src * FF + h * DKK + hl * 8);
    const v8f z = {0.f, 0.f, 0.f, 0.f, 0.f, 0.f, 0.f, 0.f};
    const v8f c0 = wmma_bf16(A, B0, z);
    const v8f c1 = wmma_bf16(A, B1, z);
#pragma unroll
    for (int j = 0; j < 8; ++j) {
      const int m = j + hl * 8;
      const int dm = __shfl(dst, m, 32);
      const float wm = __shfl(w, m, 32);
      float* arow = aggp + (size_t)dm * FF + h * DKK;
      atomicAdd(&arow[l15], wm * c0[j]);
      atomicAdd(&arow[16 + l15], wm * c1[j]);
    }
  }
}

// ---------------------------------------------------------------------------
// G = bf16(gelu(agg / (den + 1e-16)))   (exact erf gelu, per reference)
__global__ void hgt_gelu(const float* __restrict__ agg,
                         const float* __restrict__ den,
                         __bf16* __restrict__ Gb) {
  const long long total = (long long)2 * NN * FF;
  const long long st = (long long)gridDim.x * blockDim.x;
  for (long long i = (long long)blockIdx.x * blockDim.x + threadIdx.x; i < total;
       i += st) {
    const int col = (int)(i & 127);
    const long long node = i >> 7;                    // t*N + n
    const float dv = den[node * HH + (col >> 5)];
    const float v = agg[i] / (dv + 1e-16f);
    const float g = 0.5f * v * (1.0f + erff(v * 0.70710678118654752f));
    Gb[i] = (__bf16)g;
  }
}

// ---------------------------------------------------------------------------
// Output: y = G @ Wa + ba; o = alpha*y + (1-alpha)*x; LayerNorm(o). 32 rows/block.
__global__ __launch_bounds__(256) void hgt_out(
    const __bf16* __restrict__ Gb, const float* __restrict__ Wa,
    const float* __restrict__ ba, const float* __restrict__ x,
    const float* __restrict__ skip, const float* __restrict__ lns,
    const float* __restrict__ lnb, float* __restrict__ out) {
  __shared__ __bf16 wT[FF * FF];                      // 32 KB
  __shared__ float yb[32 * FF];                       // 16 KB
  const int t = blockIdx.y;
  const int rowBase = blockIdx.x * 32;
  const int wv = threadIdx.x >> 5, lane = threadIdx.x & 31;
  const int hl = lane >> 4, l15 = lane & 15;
  const int kA = hl * 8;
  const int fcol = wv * 16 + l15;
  const float* Wsrc = Wa + (size_t)t * FF * FF;
  for (int idx = threadIdx.x; idx < FF * FF; idx += 256) {
    const int k = idx >> 7, f = idx & 127;
    wT[f * FF + k] = (__bf16)Wsrc[idx];
  }
  __syncthreads();
  v16bf Bf[4];
#pragma unroll
  for (int ks = 0; ks < 4; ++ks)
    Bf[ks] = b_frag(&wT[fcol * FF + ks * 32 + hl * 8]);
  const float bias = ba[t * FF + fcol];
  const __bf16* g0 = Gb + ((size_t)t * NN + rowBase) * FF;
#pragma unroll
  for (int rt = 0; rt < 2; ++rt) {
    const __bf16* ar = g0 + (size_t)(rt * 16 + l15) * FF;
    v8f acc = {0.f, 0.f, 0.f, 0.f, 0.f, 0.f, 0.f, 0.f};
#pragma unroll
    for (int ks = 0; ks < 4; ++ks)
      acc = wmma_bf16(a_frag_bf16(ar + ks * 32 + kA), Bf[ks], acc);
#pragma unroll
    for (int j = 0; j < 8; ++j)
      yb[(rt * 16 + j + hl * 8) * FF + fcol] = acc[j] + bias;
  }
  __syncthreads();
  // Skip-mix + LayerNorm: 8 threads per row, 16 elements each, shfl reductions.
  const float alpha = 1.0f / (1.0f + expf(-skip[t]));
  const float beta = 1.0f - alpha;
  const int r = threadIdx.x >> 3;
  const int sub = threadIdx.x & 7;
  const float* xr = x + ((size_t)t * NN + rowBase + r) * FF + sub * 16;
  const float* yr = yb + r * FF + sub * 16;
  float ov[16];
  float s1 = 0.f;
#pragma unroll
  for (int i = 0; i < 16; ++i) {
    const float o = alpha * yr[i] + beta * xr[i];
    ov[i] = o;
    s1 += o;
  }
  s1 += __shfl_xor(s1, 1, 32);
  s1 += __shfl_xor(s1, 2, 32);
  s1 += __shfl_xor(s1, 4, 32);
  const float mean = s1 * (1.0f / 128.0f);
  float s2 = 0.f;
#pragma unroll
  for (int i = 0; i < 16; ++i) {
    const float dd = ov[i] - mean;
    s2 += dd * dd;
  }
  s2 += __shfl_xor(s2, 1, 32);
  s2 += __shfl_xor(s2, 2, 32);
  s2 += __shfl_xor(s2, 4, 32);
  const float rstd = rsqrtf(s2 * (1.0f / 128.0f) + 1e-5f);
  float* orow = out + ((size_t)t * NN + rowBase + r) * FF + sub * 16;
  const float* sc = lns + t * FF + sub * 16;
  const float* bi = lnb + t * FF + sub * 16;
#pragma unroll
  for (int i = 0; i < 16; ++i)
    orow[i] = (ov[i] - mean) * rstd * sc[i] + bi[i];
}

// ---------------------------------------------------------------------------
extern "C" void kernel_launch(void* const* d_in, const int* in_sizes, int n_in,
                              void* d_out, int out_size, void* d_ws,
                              size_t ws_size, hipStream_t stream) {
  const float* x = (const float*)d_in[0];
  const int* eidx = (const int*)d_in[1];
  const float* Wk = (const float*)d_in[2];
  const float* bk = (const float*)d_in[3];
  const float* Wq = (const float*)d_in[4];
  const float* bq = (const float*)d_in[5];
  const float* Wm = (const float*)d_in[6];
  const float* bm = (const float*)d_in[7];
  const float* Wa = (const float*)d_in[8];
  const float* ba = (const float*)d_in[9];
  const float* w_att = (const float*)d_in[10];
  const float* w_msg = (const float*)d_in[11];
  const float* mu = (const float*)d_in[12];
  const float* skip = (const float*)d_in[13];
  const float* lns = (const float*)d_in[14];
  const float* lnb = (const float*)d_in[15];
  (void)in_sizes; (void)n_in; (void)out_size; (void)ws_size;

  char* ws = (char*)d_ws;
  const size_t szBf = (size_t)2 * NN * FF * sizeof(__bf16);   // 102.4 MB
  const size_t off_Kb = 0;
  const size_t off_Qb = off_Kb + szBf;
  const size_t off_Mb = off_Qb + szBf;
  const size_t off_Gb = off_Mb + szBf;
  const size_t off_att = off_Gb + szBf;                        // 32 MB
  const size_t off_agg = off_att + (size_t)2 * EE * HH * 4;    // 204.8 MB
  const size_t off_den = off_agg + (size_t)2 * NN * FF * 4;    // 6.4 MB
  const size_t off_smx = off_den + (size_t)2 * NN * HH * 4;    // 6.4 MB

  __bf16* Kb = (__bf16*)(ws + off_Kb);
  __bf16* Qb = (__bf16*)(ws + off_Qb);
  __bf16* Mb = (__bf16*)(ws + off_Mb);
  __bf16* Gb = (__bf16*)(ws + off_Gb);
  float* att = (float*)(ws + off_att);
  float* agg = (float*)(ws + off_agg);
  float* den = (float*)(ws + off_den);
  unsigned int* smx = (unsigned int*)(ws + off_smx);

  // Zero agg + den + segmax (contiguous) — required each call (atomics accumulate).
  const long long zero_n16 =
      ((size_t)2 * NN * FF * 4 + (size_t)2 * NN * HH * 8) / 16;
  hgt_zero<<<2048, 256, 0, stream>>>((uint4*)(ws + off_agg), zero_n16);

  hgt_proj<<<dim3(NN / 64, 2), 256, 0, stream>>>(x, Wk, bk, Wq, bq, Wm, bm, Kb,
                                                 Qb, Mb);
  hgt_edge_att<<<dim3(1024, HH, 2), 256, 0, stream>>>(Kb, Qb, eidx, w_att, mu,
                                                      att, smx);
  hgt_edge_agg<<<dim3(1024, HH, 2), 256, 0, stream>>>(Mb, eidx, w_msg, att, smx,
                                                      den, agg);
  hgt_gelu<<<2048, 256, 0, stream>>>(agg, den, Gb);
  hgt_out<<<dim3(NN / 32, 2), 256, 0, stream>>>(Gb, Wa, ba, x, skip, lns, lnb,
                                                (float*)d_out);
}
